// MultiHeadAttention_74397423501433
// MI455X (gfx1250) — matrix operand order repaired and checked by reading, not compiled, not run on hardware
//
#include <hip/hip_runtime.h>

typedef __attribute__((ext_vector_type(16))) _Float16 v16h;
typedef __attribute__((ext_vector_type(8)))  _Float16 v8h;
typedef __attribute__((ext_vector_type(8)))  float    v8f;
typedef __attribute__((ext_vector_type(4)))  unsigned int u32x4;
typedef __attribute__((ext_vector_type(8)))  int i32x8;
typedef __attribute__((ext_vector_type(4)))  int i32x4;

#define BQ 2
#define SQ 2048
#define DQ 1024
#define HQ 16
#define DEPTH 64

static __device__ __forceinline__ v16h combine16(v8h lo, v8h hi) {
  v16h r;
#pragma unroll
  for (int i = 0; i < 8; ++i) { r[i] = lo[i]; r[i + 8] = hi[i]; }
  return r;
}

static __device__ __forceinline__ unsigned lds_off(const void* p) {
  // generic LDS address: low 32 bits are the LDS byte offset (ISA 10.2 aperture map)
  return (unsigned)(unsigned long long)(uintptr_t)p;
}

// ---------------------------------------------------------------------------
// TDM 2D tile load: global f16 tile (tile0 elems/row, tile1 rows, row stride
// stride0 elems) -> LDS with per-row padding (pad_interval/pad_amount codes).
// D# per CDNA5 ISA ch.8; tracked by TENSORcnt.  6-arg builtin variant.
// ---------------------------------------------------------------------------
static __device__ __forceinline__ void tdm_load_2d(
    unsigned lds_addr, const void* gptr, unsigned tile0, unsigned tile1,
    unsigned stride0, unsigned pad_interval, unsigned pad_amount) {
  const unsigned long long ga = (unsigned long long)(uintptr_t)gptr;
  u32x4 g0 = {};
  g0[0] = 1u;                                 // count=1 (valid user descriptor)
  g0[1] = lds_addr;                           // lds_addr[31:0]
  g0[2] = (unsigned)ga;                       // global_addr lo
  g0[3] = (unsigned)(ga >> 32) | (2u << 30);  // global_addr hi | type=2
  i32x8 g1 = {};
  g1[0] = (int)((1u << 16)                    // data_size = 1 -> 2 bytes
                | (1u << 20)                  // pad_enable
                | (pad_interval << 22) | (pad_amount << 25));
  g1[1] = 0;                                  // tensor_dim0 = 1<<30 (lo16 = 0)
  g1[2] = (int)0x4000u;                       // tensor_dim0 hi16; tensor_dim1 lo16=0
  g1[3] = (int)(0x4000u | (tile0 << 16));     // tensor_dim1 hi16 | tile_dim0
  g1[4] = (int)tile1;                         // tile_dim1 (tile_dim2 = 0)
  g1[5] = (int)stride0;                       // tensor_dim0_stride lo32
  g1[6] = 0;
  g1[7] = 0;
  i32x4 gz4 = {};
  i32x8 gz8 = {};
  __builtin_amdgcn_tensor_load_to_lds(g0, g1, gz4, gz4, gz8, 0);
}

// ---------------------------------------------------------------------------
// Prep: f32 -> f16 elementwise (8 per thread)
// ---------------------------------------------------------------------------
__global__ __launch_bounds__(256) void convert_kernel(
    const float* __restrict__ in, _Float16* __restrict__ out) {
  const int i = (blockIdx.x * 256 + threadIdx.x) * 8;
  v8h r;
#pragma unroll
  for (int j = 0; j < 8; ++j) r[j] = (_Float16)in[i + j];
  *(v8h*)(out + i) = r;
}

// Prep: in f32 [K][N] -> out f16 [N][K] (tiled 32x32 transpose)
__global__ __launch_bounds__(256) void transpose_convert_kernel(
    const float* __restrict__ in, _Float16* __restrict__ out, int K, int N) {
  __shared__ _Float16 t[32][33];
  const int n0 = blockIdx.x * 32, k0 = blockIdx.y * 32;
  const int tx = threadIdx.x & 31, ty = threadIdx.x >> 5;  // 32 x 8
#pragma unroll
  for (int i = 0; i < 4; ++i) {
    const int k = ty + i * 8;
    t[k][tx] = (_Float16)in[(size_t)(k0 + k) * N + n0 + tx];
  }
  __syncthreads();
#pragma unroll
  for (int i = 0; i < 4; ++i) {
    const int n = ty + i * 8;
    out[(size_t)(n0 + n) * K + k0 + tx] = t[tx][n];
  }
}

// ---------------------------------------------------------------------------
// f16 GEMM: C = A[MxK] * Bt[NxK]^T + bias.  128x128 tile, 8 waves, TDM
// double-buffered LDS staging.  QKV=true scatters f16 Q(.125x)/K/V^T.
// ---------------------------------------------------------------------------
template <bool QKV>
__global__ __launch_bounds__(256) void gemm_f16_kernel(
    const _Float16* __restrict__ A, const _Float16* __restrict__ Bt,
    const float* __restrict__ bias,
    _Float16* __restrict__ outQ, _Float16* __restrict__ outK,
    _Float16* __restrict__ outVt, float* __restrict__ outPlain,
    int M, int N, int K) {
  __shared__ _Float16 Alds[2][128 * 40];  // [row][k] stride 40 (TDM pad 32+8)
  __shared__ _Float16 Blds[2][128 * 40];  // [col][k] stride 40

  const int tid  = threadIdx.x;
  const int wave = tid >> 5;
  const int lane = tid & 31;
  const int lh   = lane >> 4;
  const int ln   = lane & 15;
  const int n0 = blockIdx.x * 128;
  const int m0 = blockIdx.y * 128;

  const v8f vzero = {};
  v8f acc[8];
#pragma unroll
  for (int t = 0; t < 8; ++t) acc[t] = vzero;

  const _Float16* Abase = A + (size_t)m0 * K;
  const _Float16* Bbase = Bt + (size_t)n0 * K;
  const int steps = K / 32;

  if (wave == 0) {  // prologue: stage tile 0 (pad: 16 dwords -> +4 dwords => stride 40)
    tdm_load_2d(lds_off(&Alds[0][0]), Abase, 32, 128, (unsigned)K, 3u, 3u);
    tdm_load_2d(lds_off(&Blds[0][0]), Bbase, 32, 128, (unsigned)K, 3u, 3u);
  }

  for (int i = 0; i < steps; ++i) {
    const int cur = i & 1;
    if (wave == 0) {
      if (i + 1 < steps) {
        const int k1 = (i + 1) * 32;
        tdm_load_2d(lds_off(&Alds[cur ^ 1][0]), Abase + k1, 32, 128, (unsigned)K, 3u, 3u);
        tdm_load_2d(lds_off(&Blds[cur ^ 1][0]), Bbase + k1, 32, 128, (unsigned)K, 3u, 3u);
        __builtin_amdgcn_s_wait_tensorcnt(2);  // tile i done, tile i+1 in flight
      } else {
        __builtin_amdgcn_s_wait_tensorcnt(0);
      }
    }
    __syncthreads();

    const _Float16* bp = &Blds[cur][(wave * 16 + ln) * 40 + lh * 8];
    const v16h bmat = combine16(*(const v8h*)bp, *(const v8h*)(bp + 16));
#pragma unroll
    for (int tm = 0; tm < 8; ++tm) {
      const _Float16* ap = &Alds[cur][(tm * 16 + ln) * 40 + lh * 8];
      const v16h amat = combine16(*(const v8h*)ap, *(const v8h*)(ap + 16));
      acc[tm] = __builtin_amdgcn_wmma_f32_16x16x32_f16(
          false, amat, false, bmat, (short)0, acc[tm], false, false);
    }
    __syncthreads();
  }

  const int n  = n0 + wave * 16 + ln;
  const float bv = bias[n];
#pragma unroll
  for (int tm = 0; tm < 8; ++tm) {
#pragma unroll
    for (int r = 0; r < 8; ++r) {
      const int m = m0 + tm * 16 + r + lh * 8;
      const float val = acc[tm][r] + bv;
      if (QKV) {
        const int part = n >> 10;
        const int rem  = n & 1023;
        const int hh   = rem >> 6;
        const int dd   = rem & 63;
        const int b    = m >> 11;
        const int s    = m & (SQ - 1);
        const int bh   = b * HQ + hh;
        if (part == 0)
          outQ[((size_t)bh * SQ + s) * DEPTH + dd] = (_Float16)(val * 0.125f);
        else if (part == 1)
          outK[((size_t)bh * SQ + s) * DEPTH + dd] = (_Float16)val;
        else
          outVt[((size_t)bh * DEPTH + dd) * SQ + s] = (_Float16)val;
      } else {
        outPlain[(size_t)m * N + n] = val;
      }
    }
  }
}

// ---------------------------------------------------------------------------
// Causal flash attention.  8 waves x 16 query rows; 64-key blocks staged by
// TDM double buffering (K row-major = B-operand layout; V pre-transposed).
// ---------------------------------------------------------------------------
__global__ __launch_bounds__(256) void attn_kernel(
    const _Float16* __restrict__ Qh, const _Float16* __restrict__ Kh,
    const _Float16* __restrict__ Vth, const float* __restrict__ mask,
    _Float16* __restrict__ ctx) {
  __shared__ _Float16 Klds[2][64 * 72];   // [key][d], stride 72 (TDM pad 32+4 dw)
  __shared__ _Float16 Vtlds[2][64 * 72];  // [d][key], stride 72
  __shared__ _Float16 Plds[8 * 16 * 72];  // per-wave P strip

  const int tid  = threadIdx.x;
  const int wave = tid >> 5;
  const int lane = tid & 31;
  const int lh   = lane >> 4;
  const int ln   = lane & 15;

  const int qblk = blockIdx.x;
  const int bh   = blockIdx.y;
  const int b    = bh >> 4;
  const int hh   = bh & 15;
  const int qbase = qblk * 128;
  const int qrow_wave = qbase + wave * 16;

  const _Float16* Kbase  = Kh + (size_t)bh * SQ * DEPTH;
  const _Float16* Vtbase = Vth + (size_t)bh * DEPTH * SQ;

  // Q fragment (already scaled by 1/sqrt(DEPTH)) straight from global f16
  const _Float16* qptr = Qh + ((size_t)bh * SQ + qrow_wave + ln) * DEPTH;
  v16h qa[2];
#pragma unroll
  for (int c = 0; c < 2; ++c)
    qa[c] = combine16(*(const v8h*)(qptr + c * 32 + lh * 8),
                      *(const v8h*)(qptr + c * 32 + lh * 8 + 16));

  const v8f vzero = {};
  v8f acc[4];
#pragma unroll
  for (int t = 0; t < 4; ++t) acc[t] = vzero;
  float mrow[8], lsum[8];
#pragma unroll
  for (int r = 0; r < 8; ++r) { mrow[r] = -3.0e38f; lsum[r] = 0.0f; }

  const int nkblocks = (qbase + 128) >> 6;

  if (wave == 0) {  // prologue tiles (pad: 32 dwords -> +4 dwords => stride 72)
    tdm_load_2d(lds_off(&Klds[0][0]), Kbase, 64, 64, DEPTH, 4u, 3u);
    tdm_load_2d(lds_off(&Vtlds[0][0]), Vtbase, 64, 64, SQ, 4u, 3u);
  }

  for (int kb = 0; kb < nkblocks; ++kb) {
    const int cur = kb & 1;
    const int kbase = kb * 64;
    if (wave == 0) {
      if (kb + 1 < nkblocks) {
        tdm_load_2d(lds_off(&Klds[cur ^ 1][0]), Kbase + (size_t)(kbase + 64) * DEPTH,
                    64, 64, DEPTH, 4u, 3u);
        tdm_load_2d(lds_off(&Vtlds[cur ^ 1][0]), Vtbase + kbase + 64,
                    64, 64, SQ, 4u, 3u);
        __builtin_amdgcn_s_wait_tensorcnt(2);
      } else {
        __builtin_amdgcn_s_wait_tensorcnt(0);
      }
    }
    __syncthreads();

    // scores: 4 key tiles x 2 depth chunks
    v8f sc[4];
#pragma unroll
    for (int kt = 0; kt < 4; ++kt) {
      v8f creg = vzero;
#pragma unroll
      for (int cc = 0; cc < 2; ++cc) {
        const _Float16* bp = &Klds[cur][(kt * 16 + ln) * 72 + cc * 32 + lh * 8];
        const v16h bm = combine16(*(const v8h*)bp, *(const v8h*)(bp + 16));
        creg = __builtin_amdgcn_wmma_f32_16x16x32_f16(
            false, qa[cc], false, bm, (short)0, creg, false, false);
      }
      sc[kt] = creg;
    }

    // additive + causal mask
#pragma unroll
    for (int kt = 0; kt < 4; ++kt) {
      const int ki = kbase + kt * 16 + ln;
      const float mv = mask[b * SQ + ki];
#pragma unroll
      for (int r = 0; r < 8; ++r) {
        const int qi = qrow_wave + r + lh * 8;
        float sv = sc[kt][r] + mv;
        if (ki > qi) sv = -1.0e9f;
        sc[kt][r] = sv;
      }
    }

    // online softmax (row stats live in 16-lane halves)
    float fac[8];
#pragma unroll
    for (int r = 0; r < 8; ++r) {
      float v = fmaxf(fmaxf(sc[0][r], sc[1][r]), fmaxf(sc[2][r], sc[3][r]));
#pragma unroll
      for (int off = 1; off < 16; off <<= 1) v = fmaxf(v, __shfl_xor(v, off, 32));
      const float mnew = fmaxf(mrow[r], v);
      fac[r] = __expf(mrow[r] - mnew);
      mrow[r] = mnew;
    }

    _Float16* pbase = &Plds[wave * 16 * 72];
    float rsum[8];
#pragma unroll
    for (int r = 0; r < 8; ++r) rsum[r] = 0.0f;
#pragma unroll
    for (int kt = 0; kt < 4; ++kt) {
#pragma unroll
      for (int r = 0; r < 8; ++r) {
        const float p = __expf(sc[kt][r] - mrow[r]);
        rsum[r] += p;
        pbase[(r + lh * 8) * 72 + kt * 16 + ln] = (_Float16)p;
      }
    }
#pragma unroll
    for (int r = 0; r < 8; ++r) {
#pragma unroll
      for (int off = 1; off < 16; off <<= 1) rsum[r] += __shfl_xor(rsum[r], off, 32);
      lsum[r] = lsum[r] * fac[r] + rsum[r];
    }
#pragma unroll
    for (int t = 0; t < 4; ++t)
#pragma unroll
      for (int r = 0; r < 8; ++r) acc[t][r] *= fac[r];

    // O += P @ V
#pragma unroll
    for (int cc = 0; cc < 2; ++cc) {
      const _Float16* ap = pbase + ln * 72 + cc * 32 + lh * 8;
      const v16h am = combine16(*(const v8h*)ap, *(const v8h*)(ap + 16));
#pragma unroll
      for (int t = 0; t < 4; ++t) {
        const _Float16* bp = &Vtlds[cur][(t * 16 + ln) * 72 + cc * 32 + lh * 8];
        const v16h bm = combine16(*(const v8h*)bp, *(const v8h*)(bp + 16));
        acc[t] = __builtin_amdgcn_wmma_f32_16x16x32_f16(
            false, am, false, bm, (short)0, acc[t], false, false);
      }
    }
    __syncthreads();
  }

  // normalize + store merged-head ctx (f16 feeds final TDM GEMM)
#pragma unroll
  for (int r = 0; r < 8; ++r) {
    const float inv = 1.0f / lsum[r];
    const int s = qrow_wave + r + lh * 8;
    _Float16* dst = ctx + (size_t)(b * SQ + s) * DQ + hh * DEPTH;
#pragma unroll
    for (int t = 0; t < 4; ++t) dst[t * 16 + ln] = (_Float16)(acc[t][r] * inv);
  }
}

// ---------------------------------------------------------------------------
extern "C" void kernel_launch(void* const* d_in, const int* in_sizes, int n_in,
                              void* d_out, int out_size, void* d_ws, size_t ws_size,
                              hipStream_t stream) {
  const float* token = (const float*)d_in[0];
  const float* mask  = (const float*)d_in[1];
  const float* Wqkv  = (const float*)d_in[2];
  const float* bqkv  = (const float*)d_in[3];
  const float* Wout  = (const float*)d_in[4];
  const float* bout  = (const float*)d_in[5];
  float* out = (float*)d_out;

  _Float16* ws = (_Float16*)d_ws;
  const size_t MT  = (size_t)BQ * SQ * DQ;  // 4M elements
  _Float16* Xh     = ws;                    // [B*S][D]
  _Float16* Wqkvt  = Xh + MT;               // [3D][D]
  _Float16* Woutt  = Wqkvt + 3 * (size_t)DQ * DQ;
  _Float16* qh     = Woutt + (size_t)DQ * DQ;  // [bh][s][d], pre-scaled
  _Float16* kh     = qh + MT;                  // [bh][s][d]
  _Float16* vth    = kh + MT;                  // [bh][d][s]
  _Float16* ctxh   = vth + MT;                 // [B*S][D]

  // prep: one-time convert/transpose so all GEMM operands are TDM-ready f16
  convert_kernel<<<dim3(MT / (256 * 8)), 256, 0, stream>>>(token, Xh);
  transpose_convert_kernel<<<dim3(3 * DQ / 32, DQ / 32), 256, 0, stream>>>(
      Wqkv, Wqkvt, DQ, 3 * DQ);
  transpose_convert_kernel<<<dim3(DQ / 32, DQ / 32), 256, 0, stream>>>(
      Wout, Woutt, DQ, DQ);

  // 1) fused QKV projection -> head-split f16 Q(.125x)/K/V^T
  gemm_f16_kernel<true><<<dim3(3 * DQ / 128, BQ * SQ / 128), 256, 0, stream>>>(
      Xh, Wqkvt, bqkv, qh, kh, vth, nullptr, BQ * SQ, 3 * DQ, DQ);
  // 2) causal flash attention -> merged-head ctx (f16)
  attn_kernel<<<dim3(SQ / 128, BQ * HQ), 256, 0, stream>>>(qh, kh, vth, mask, ctxh);
  // 3) output projection -> f32 out
  gemm_f16_kernel<false><<<dim3(DQ / 128, BQ * SQ / 128), 256, 0, stream>>>(
      ctxh, Woutt, bout, nullptr, nullptr, nullptr, out, BQ * SQ, DQ, DQ);
}
